// MoE_26474178412922
// MI455X (gfx1250) — compile-verified, matrix operand order checked by reading, not run on hardware
//
#include <hip/hip_runtime.h>

// ---------------- problem constants ----------------
#define D_DIM   1024
#define S_DIM   2048
#define B_DIM   4
#define H_DIM   4096
#define NHEADS  16
#define HD      64
#define E_DIM   8
#define C_CAP   512
#define NTOK    (B_DIM * S_DIM)      // 8192
#define LD3     (3 * D_DIM)          // 3072

typedef __attribute__((ext_vector_type(16))) _Float16 v16h;
typedef __attribute__((ext_vector_type(8)))  float    v8f;

// fp32 source, contiguous in K (16B aligned), convert to fp16 fragment
__device__ __forceinline__ v16h load_frag_f32_contig(const float* __restrict__ p, int half) {
    const float4* q0 = reinterpret_cast<const float4*>(p + half * 8);
    const float4* q1 = reinterpret_cast<const float4*>(p + 16 + half * 8);
    float4 a = q0[0], b = q0[1], c = q1[0], d = q1[1];
    v16h f;
    f[0]  = (_Float16)a.x; f[1]  = (_Float16)a.y; f[2]  = (_Float16)a.z; f[3]  = (_Float16)a.w;
    f[4]  = (_Float16)b.x; f[5]  = (_Float16)b.y; f[6]  = (_Float16)b.z; f[7]  = (_Float16)b.w;
    f[8]  = (_Float16)c.x; f[9]  = (_Float16)c.y; f[10] = (_Float16)c.z; f[11] = (_Float16)c.w;
    f[12] = (_Float16)d.x; f[13] = (_Float16)d.y; f[14] = (_Float16)d.z; f[15] = (_Float16)d.w;
    return f;
}

// fp16 source in LDS, contiguous in K
__device__ __forceinline__ v16h load_frag_h_contig(const _Float16* p, int half) {
    v16h f;
#pragma unroll
    for (int i = 0; i < 8; ++i) f[i] = p[half * 8 + i];
#pragma unroll
    for (int i = 0; i < 8; ++i) f[8 + i] = p[16 + half * 8 + i];
    return f;
}

__device__ __forceinline__ void cvt_store4(_Float16* d, float4 v) {
    d[0] = (_Float16)v.x; d[1] = (_Float16)v.y;
    d[2] = (_Float16)v.z; d[3] = (_Float16)v.w;
}

// ---------------- WMMA GEMM: C = act(A @ B + bias) ----------------
// A: [M,K] row-major (lda), batch = blockIdx.z with strideA.
// B: element (k,n) at Bm[k*ldbk + n*ldbn], batch (z % bmod)*strideB.
//    BKC=true  -> K-contiguous rows (weight^T layout [N,K])
//    BKC=false -> N-contiguous rows (weight layout [K,N]); transposed into LDS.
// Block 256 threads = 8 waves (4 M-subtiles x 2 N-subtiles); wave tile 32x64
// (two M halves); block tile 128x128. Software pipeline: global->regs for the
// next k-tile overlapped with WMMA on the current LDS tile; k+2 prefetched.
template <bool BKC, bool RELU>
__global__ __launch_bounds__(256) void gemm_wmma_kernel(
    const float* __restrict__ A, long long lda, long long strideA,
    const float* __restrict__ Bm, long long ldbk, long long ldbn,
    long long strideB, int bmod,
    const float* __restrict__ bias,
    float* __restrict__ Cp, long long ldc, long long strideC,
    int M, int N, int K)
{
    __shared__ _Float16 As[2][128 * 32];   // [row][k]
    __shared__ _Float16 Bs[2][128 * 32];   // [n][k]

    int z = blockIdx.z;
    const float* Ab = A + (long long)z * strideA;
    const float* Bb = Bm + (long long)(z % bmod) * strideB;
    float* Cb = Cp + (long long)z * strideC;

    int tid  = threadIdx.x;
    int wave = tid >> 5, lane = tid & 31;
    int half = lane >> 4, r16 = lane & 15;
    int wm = wave & 3, wn = wave >> 2;
    int bm0 = blockIdx.y * 128;
    int bn0 = blockIdx.x * 128;
    int nk = K >> 5;

    float4 aReg[4];
    float4 bReg[4];

    auto gload = [&](int k0) {
#pragma unroll
        for (int i = 0; i < 4; ++i) {
            int idx = tid + i * 256;              // 0..1023
            int row = idx >> 3, q = idx & 7;
            aReg[i] = *reinterpret_cast<const float4*>(
                Ab + (long long)(bm0 + row) * lda + k0 + q * 4);
        }
#pragma unroll
        for (int i = 0; i < 4; ++i) {
            int idx = tid + i * 256;              // 0..1023
            if (BKC) {
                int n = idx >> 3, q = idx & 7;
                bReg[i] = *reinterpret_cast<const float4*>(
                    Bb + (long long)(bn0 + n) * ldbn + k0 + q * 4);
            } else {
                int k = idx >> 5, nq = idx & 31;
                bReg[i] = *reinterpret_cast<const float4*>(
                    Bb + (long long)(k0 + k) * ldbk + bn0 + nq * 4);
            }
        }
    };
    auto lstore = [&](int buf) {
#pragma unroll
        for (int i = 0; i < 4; ++i) {
            int idx = tid + i * 256;
            int row = idx >> 3, q = idx & 7;
            cvt_store4(&As[buf][row * 32 + q * 4], aReg[i]);
        }
#pragma unroll
        for (int i = 0; i < 4; ++i) {
            int idx = tid + i * 256;
            if (BKC) {
                int n = idx >> 3, q = idx & 7;
                cvt_store4(&Bs[buf][n * 32 + q * 4], bReg[i]);
            } else {
                int k = idx >> 5, nq = idx & 31;
                Bs[buf][(nq * 4 + 0) * 32 + k] = (_Float16)bReg[i].x;
                Bs[buf][(nq * 4 + 1) * 32 + k] = (_Float16)bReg[i].y;
                Bs[buf][(nq * 4 + 2) * 32 + k] = (_Float16)bReg[i].z;
                Bs[buf][(nq * 4 + 3) * 32 + k] = (_Float16)bReg[i].w;
            }
        }
    };

    v8f acc0[4] = {};
    v8f acc1[4] = {};
    gload(0);
    int buf = 0;
    for (int ks = 0; ks < nk; ++ks) {
        lstore(buf);
        __syncthreads();
        if (ks + 1 < nk) gload((ks + 1) << 5);       // hide latency behind WMMA
        if (ks + 2 < nk) {                            // gfx1250 global_prefetch_b8
            int k0p = (ks + 2) << 5;
            __builtin_prefetch(Ab + (long long)(bm0 + (tid >> 3)) * lda + k0p + (tid & 7) * 4, 0, 1);
            if (BKC)
                __builtin_prefetch(Bb + (long long)(bn0 + (tid >> 3)) * ldbn + k0p + (tid & 7) * 4, 0, 1);
            else
                __builtin_prefetch(Bb + (long long)(k0p + (tid >> 5)) * ldbk + bn0 + (tid & 31) * 4, 0, 1);
        }
        v16h af0 = load_frag_h_contig(&As[buf][(wm * 16 + r16) * 32], half);
        v16h af1 = load_frag_h_contig(&As[buf][(64 + wm * 16 + r16) * 32], half);
#pragma unroll
        for (int t = 0; t < 4; ++t) {
            const _Float16* brow = &Bs[buf][(wn * 64 + t * 16 + r16) * 32];
            v16h bf = load_frag_h_contig(brow, half);
            acc0[t] = __builtin_amdgcn_wmma_f32_16x16x32_f16(
                false, af0, false, bf, (short)0, acc0[t], false, false);
            acc1[t] = __builtin_amdgcn_wmma_f32_16x16x32_f16(
                false, af1, false, bf, (short)0, acc1[t], false, false);
        }
        __syncthreads();
        buf ^= 1;
    }

#pragma unroll
    for (int t = 0; t < 4; ++t) {
#pragma unroll
        for (int r = 0; r < 8; ++r) {
            int n = bn0 + wn * 64 + t * 16 + r16;
            float bi = bias ? bias[n] : 0.0f;
            int m0 = bm0 + wm * 16 + r + half * 8;
            float v0 = acc0[t][r] + bi;
            if (RELU) v0 = fmaxf(v0, 0.0f);
            Cb[(long long)m0 * ldc + n] = v0;
            int m1 = m0 + 64;
            float v1 = acc1[t][r] + bi;
            if (RELU) v1 = fmaxf(v1, 0.0f);
            Cb[(long long)m1 * ldc + n] = v1;
        }
    }
}

// ---------------- Flash attention (wave32 WMMA, LDS-staged K/V) ----------------
// qkv: [B, S, 3D] packed Q|K|V per token. out: [B, S, D] heads merged.
// Block = 128 threads = 4 waves; each wave owns 16 query rows; 32 keys/step.
__global__ __launch_bounds__(128) void attention_kernel(
    const float* __restrict__ qkv, float* __restrict__ out)
{
    __shared__ _Float16 Kbuf[32][64];     // [key][d]
    __shared__ _Float16 Vbuf[64][32];     // [d][key]  (transposed at staging)
    __shared__ _Float16 Pb[4][16][32];    // per-wave softmax tile

    int b = blockIdx.z, h = blockIdx.y;
    int tid  = threadIdx.x;
    int wave = tid >> 5, lane = tid & 31;
    int half = lane >> 4, r16 = lane & 15;
    int q0 = blockIdx.x * 64 + wave * 16;

    const float* base = qkv + (long long)b * S_DIM * LD3;
    const float* Qp = base + h * HD;
    const float* Kp = base + D_DIM + h * HD;
    const float* Vp = base + 2 * D_DIM + h * HD;

    const float* qrow = Qp + (long long)(q0 + r16) * LD3;
    v16h qf0 = load_frag_f32_contig(qrow, half);
    v16h qf1 = load_frag_f32_contig(qrow + 32, half);

    v8f accO[4] = {};
    float mrow[8], lrow[8];
#pragma unroll
    for (int r = 0; r < 8; ++r) { mrow[r] = -1e30f; lrow[r] = 0.0f; }

    const float scale = 0.125f; // 1/sqrt(64)

    for (int kv0 = 0; kv0 < S_DIM; kv0 += 32) {
        // ---- cooperative staging: K tile [32][64], V tile transposed [64][32]
#pragma unroll
        for (int i = 0; i < 4; ++i) {
            int idx = tid + i * 128;              // 0..511
            int key = idx >> 4, dq = idx & 15;
            float4 kv4 = *reinterpret_cast<const float4*>(
                Kp + (long long)(kv0 + key) * LD3 + dq * 4);
            cvt_store4(&Kbuf[key][dq * 4], kv4);
            float4 vv4 = *reinterpret_cast<const float4*>(
                Vp + (long long)(kv0 + key) * LD3 + dq * 4);
            Vbuf[dq * 4 + 0][key] = (_Float16)vv4.x;
            Vbuf[dq * 4 + 1][key] = (_Float16)vv4.y;
            Vbuf[dq * 4 + 2][key] = (_Float16)vv4.z;
            Vbuf[dq * 4 + 3][key] = (_Float16)vv4.w;
            if (kv0 + 32 < S_DIM) {
                __builtin_prefetch(Kp + (long long)(kv0 + 32 + key) * LD3 + dq * 4, 0, 1);
                __builtin_prefetch(Vp + (long long)(kv0 + 32 + key) * LD3 + dq * 4, 0, 1);
            }
        }
        __syncthreads();

        // ---- scores: Q(16x64) @ K^T(64x32) -> two 16x16 tiles
        v8f sc[2];
#pragma unroll
        for (int t = 0; t < 2; ++t) {
            const _Float16* krow = &Kbuf[t * 16 + r16][0];
            v16h kf0 = load_frag_h_contig(krow, half);
            v16h kf1 = load_frag_h_contig(krow + 32, half);
            v8f zacc = {};
            zacc = __builtin_amdgcn_wmma_f32_16x16x32_f16(
                false, qf0, false, kf0, (short)0, zacc, false, false);
            zacc = __builtin_amdgcn_wmma_f32_16x16x32_f16(
                false, qf1, false, kf1, (short)0, zacc, false, false);
            sc[t] = zacc;
        }

        // ---- online softmax; each query row spans the 16 lanes of this half
#pragma unroll
        for (int r = 0; r < 8; ++r) {
            float s0 = sc[0][r] * scale;
            float s1 = sc[1][r] * scale;
            float mx = fmaxf(s0, s1);
#pragma unroll
            for (int msk = 1; msk <= 8; msk <<= 1)
                mx = fmaxf(mx, __shfl_xor(mx, msk, 32));
            float mnew = fmaxf(mrow[r], mx);
            float corr = __expf(mrow[r] - mnew);
            float p0 = __expf(s0 - mnew);
            float p1 = __expf(s1 - mnew);
            float ps = p0 + p1;
#pragma unroll
            for (int msk = 1; msk <= 8; msk <<= 1)
                ps += __shfl_xor(ps, msk, 32);
            lrow[r] = lrow[r] * corr + ps;
            mrow[r] = mnew;
#pragma unroll
            for (int t = 0; t < 4; ++t) accO[t][r] *= corr;
            int mr = r + half * 8;
            Pb[wave][mr][r16]      = (_Float16)p0;
            Pb[wave][mr][16 + r16] = (_Float16)p1;
        }
        __syncthreads();

        // ---- O += P(16x32) @ V(32x64)
        v16h pf = load_frag_h_contig(&Pb[wave][r16][0], half);
#pragma unroll
        for (int t = 0; t < 4; ++t) {
            v16h vf = load_frag_h_contig(&Vbuf[t * 16 + r16][0], half);
            accO[t] = __builtin_amdgcn_wmma_f32_16x16x32_f16(
                false, pf, false, vf, (short)0, accO[t], false, false);
        }
        __syncthreads();   // protect Kbuf/Vbuf before next staging
    }

    float* ob = out + (long long)b * S_DIM * D_DIM + h * HD;
#pragma unroll
    for (int t = 0; t < 4; ++t) {
#pragma unroll
        for (int r = 0; r < 8; ++r) {
            int q = q0 + r + half * 8;
            ob[(long long)q * D_DIM + t * 16 + r16] = accO[t][r] / lrow[r];
        }
    }
}

// ---------------- residual add + LayerNorm ----------------
__global__ __launch_bounds__(256) void add_ln_kernel(
    const float* __restrict__ x, const float* __restrict__ res,
    const float* __restrict__ w, const float* __restrict__ bvec,
    float* __restrict__ out)
{
    int row = blockIdx.x;
    const float* xr = x   + (long long)row * D_DIM;
    const float* rr = res + (long long)row * D_DIM;
    float v[4], s = 0.0f, s2 = 0.0f;
#pragma unroll
    for (int i = 0; i < 4; ++i) {
        int idx = threadIdx.x + i * 256;
        float t = xr[idx] + rr[idx];
        v[i] = t; s += t; s2 += t * t;
    }
    __shared__ float sh1[256], sh2[256];
    sh1[threadIdx.x] = s; sh2[threadIdx.x] = s2;
    __syncthreads();
    for (int off = 128; off > 0; off >>= 1) {
        if (threadIdx.x < off) {
            sh1[threadIdx.x] += sh1[threadIdx.x + off];
            sh2[threadIdx.x] += sh2[threadIdx.x + off];
        }
        __syncthreads();
    }
    float mean = sh1[0] * (1.0f / D_DIM);
    float var  = sh2[0] * (1.0f / D_DIM) - mean * mean;
    float rstd = rsqrtf(var + 1e-5f);
    float* orow = out + (long long)row * D_DIM;
#pragma unroll
    for (int i = 0; i < 4; ++i) {
        int idx = threadIdx.x + i * 256;
        orow[idx] = (v[i] - mean) * rstd * w[idx] + bvec[idx];
    }
}

// ---------------- switch gating: top-1 of softmax(x @ gate_w) ----------------
__global__ __launch_bounds__(256) void gate_kernel(
    const float* __restrict__ x, const float* __restrict__ gw,
    float* __restrict__ gateval, int* __restrict__ gateidx)
{
    int tok = blockIdx.x;
    const float* xr = x + (long long)tok * D_DIM;
    float acc[E_DIM] = {};
    for (int d = threadIdx.x; d < D_DIM; d += 256) {
        float xv = xr[d];
#pragma unroll
        for (int e = 0; e < E_DIM; ++e) acc[e] += xv * gw[d * E_DIM + e];
    }
    __shared__ float sh[256];
    __shared__ float logits[E_DIM];
    for (int e = 0; e < E_DIM; ++e) {
        sh[threadIdx.x] = acc[e];
        __syncthreads();
        for (int off = 128; off > 0; off >>= 1) {
            if (threadIdx.x < off) sh[threadIdx.x] += sh[threadIdx.x + off];
            __syncthreads();
        }
        if (threadIdx.x == 0) logits[e] = sh[0];
        __syncthreads();
    }
    if (threadIdx.x == 0) {
        float lmax = logits[0]; int imax = 0;
        for (int e = 1; e < E_DIM; ++e)
            if (logits[e] > lmax) { lmax = logits[e]; imax = e; }
        float den = 0.0f;
        for (int e = 0; e < E_DIM; ++e) den += __expf(logits[e] - lmax);
        gateval[tok] = 1.0f / den;   // max of softmax
        gateidx[tok] = imax;
    }
}

// one thread per (b,e): sequential capacity scan (1-based pos, keep if pos < C)
__global__ void route_scan_kernel(const int* __restrict__ eidx, int* __restrict__ slot)
{
    int t = threadIdx.x + blockIdx.x * blockDim.x;
    if (t >= B_DIM * E_DIM) return;
    int b = t / E_DIM, e = t % E_DIM;
    int count = 0;
    for (int s = 0; s < S_DIM; ++s) {
        int tok = b * S_DIM + s;
        if (eidx[tok] == e) {
            ++count;
            slot[tok] = (count < C_CAP) ? count : -1;
        }
    }
}

__global__ void fill_f32_kernel(float* p, long long n, float v)
{
    long long i = (long long)blockIdx.x * blockDim.x + threadIdx.x;
    long long stride = (long long)gridDim.x * blockDim.x;
    for (; i < n; i += stride) p[i] = v;
}

__global__ void fill_i32_kernel(int* p, long long n, int v)
{
    long long i = (long long)blockIdx.x * blockDim.x + threadIdx.x;
    long long stride = (long long)gridDim.x * blockDim.x;
    for (; i < n; i += stride) p[i] = v;
}

__global__ __launch_bounds__(256) void gather_kernel(
    const float* __restrict__ x, const int* __restrict__ eidx,
    const int* __restrict__ slot, float* __restrict__ expin)
{
    int tok = blockIdx.x;
    int s = slot[tok];
    if (s < 0) return;
    int b = tok / S_DIM;
    int e = eidx[tok];
    const float4* src = reinterpret_cast<const float4*>(x + (long long)tok * D_DIM);
    float4* dst = reinterpret_cast<float4*>(
        expin + ((((long long)b * E_DIM + e) * C_CAP) + s) * D_DIM);
    dst[threadIdx.x] = src[threadIdx.x];
}

__global__ __launch_bounds__(256) void combine_kernel(
    const float* __restrict__ expout, const float* __restrict__ gateval,
    const int* __restrict__ eidx, const int* __restrict__ slot,
    float* __restrict__ moe)
{
    int tok = blockIdx.x;
    int s = slot[tok];
    float4* dst = reinterpret_cast<float4*>(moe + (long long)tok * D_DIM);
    if (s < 0) { dst[threadIdx.x] = make_float4(0.f, 0.f, 0.f, 0.f); return; }
    int b = tok / S_DIM;
    int e = eidx[tok];
    float g = gateval[tok];
    const float4* src = reinterpret_cast<const float4*>(
        expout + ((((long long)b * E_DIM + e) * C_CAP) + s) * D_DIM);
    float4 v = src[threadIdx.x];
    dst[threadIdx.x] = make_float4(v.x * g, v.y * g, v.z * g, v.w * g);
}

__global__ __launch_bounds__(256) void rowsum_kernel(
    const float* __restrict__ x, float* __restrict__ part)
{
    int row = blockIdx.x;
    const float* xr = x + (long long)row * D_DIM;
    float s = 0.0f;
    for (int i = threadIdx.x; i < D_DIM; i += 256) s += xr[i];
    __shared__ float sh[256];
    sh[threadIdx.x] = s;
    __syncthreads();
    for (int off = 128; off > 0; off >>= 1) {
        if (threadIdx.x < off) sh[threadIdx.x] += sh[threadIdx.x + off];
        __syncthreads();
    }
    if (threadIdx.x == 0) part[row] = sh[0];
}

__global__ __launch_bounds__(256) void final_reduce_kernel(
    const float* __restrict__ part, float* __restrict__ out)
{
    float s = 0.0f;
    for (int i = threadIdx.x; i < NTOK; i += 256) s += part[i];
    __shared__ float sh[256];
    sh[threadIdx.x] = s;
    __syncthreads();
    for (int off = 128; off > 0; off >>= 1) {
        if (threadIdx.x < off) sh[threadIdx.x] += sh[threadIdx.x + off];
        __syncthreads();
    }
    if (threadIdx.x == 0) out[0] = sh[0];
}

// ---------------- host-side GEMM dispatch ----------------
static void launch_gemm(bool bkc, bool relu,
                        const float* A, long long lda, long long sA,
                        const float* Bm, long long ldbk, long long ldbn,
                        long long sB, int bmod,
                        const float* bias,
                        float* C, long long ldc, long long sC,
                        int M, int N, int K, int batches, hipStream_t st)
{
    dim3 grid(N / 128, M / 128, batches), block(256);
    if (bkc) {
        if (relu) gemm_wmma_kernel<true, true><<<grid, block, 0, st>>>(
            A, lda, sA, Bm, ldbk, ldbn, sB, bmod, bias, C, ldc, sC, M, N, K);
        else      gemm_wmma_kernel<true, false><<<grid, block, 0, st>>>(
            A, lda, sA, Bm, ldbk, ldbn, sB, bmod, bias, C, ldc, sC, M, N, K);
    } else {
        if (relu) gemm_wmma_kernel<false, true><<<grid, block, 0, st>>>(
            A, lda, sA, Bm, ldbk, ldbn, sB, bmod, bias, C, ldc, sC, M, N, K);
        else      gemm_wmma_kernel<false, false><<<grid, block, 0, st>>>(
            A, lda, sA, Bm, ldbk, ldbn, sB, bmod, bias, C, ldc, sC, M, N, K);
    }
}

// ---------------- workspace layout (float offsets) ----------------
#define OFF_BIG   0LL                          // 67,108,864 floats (qkv / ffn-h / expert-h)
#define OFF_R1    67108864LL                   // 8,388,608 each
#define OFF_R2    75497472LL
#define OFF_R3    83886080LL
#define OFF_R4    92274688LL
#define OFF_EXP   100663296LL                  // 16,777,216
#define OFF_GATE  117440512LL                  // 8192
#define OFF_PART  117448704LL                  // 8192
#define OFF_IDX   117456896LL                  // 8192 ints
#define OFF_SLOT  117465088LL                  // 8192 ints
#define WS_FLOATS 117473280LL

extern "C" void kernel_launch(void* const* d_in, const int* in_sizes, int n_in,
                              void* d_out, int out_size, void* d_ws, size_t ws_size,
                              hipStream_t stream)
{
    if (ws_size < (size_t)WS_FLOATS * sizeof(float)) return;

    const float* x        = (const float*)d_in[0];
    const float* l0_in_w  = (const float*)d_in[1];
    const float* l0_in_b  = (const float*)d_in[2];
    const float* l0_out_w = (const float*)d_in[3];
    const float* l0_out_b = (const float*)d_in[4];
    const float* l0_l1_w  = (const float*)d_in[5];
    const float* l0_l1_b  = (const float*)d_in[6];
    const float* l0_l2_w  = (const float*)d_in[7];
    const float* l0_l2_b  = (const float*)d_in[8];
    const float* l0_n1_w  = (const float*)d_in[9];
    const float* l0_n1_b  = (const float*)d_in[10];
    const float* l0_n2_w  = (const float*)d_in[11];
    const float* l0_n2_b  = (const float*)d_in[12];
    const float* l1_in_w  = (const float*)d_in[13];
    const float* l1_in_b  = (const float*)d_in[14];
    const float* l1_out_w = (const float*)d_in[15];
    const float* l1_out_b = (const float*)d_in[16];
    const float* l1_gw    = (const float*)d_in[17];
    const float* l1_w1    = (const float*)d_in[18];
    const float* l1_w2    = (const float*)d_in[19];
    const float* l1_n1_w  = (const float*)d_in[20];
    const float* l1_n1_b  = (const float*)d_in[21];
    const float* l1_n2_w  = (const float*)d_in[22];
    const float* l1_n2_b  = (const float*)d_in[23];

    float* ws    = (float*)d_ws;
    float* big   = ws + OFF_BIG;
    float* r1    = ws + OFF_R1;
    float* r2    = ws + OFF_R2;
    float* r3    = ws + OFF_R3;
    float* r4    = ws + OFF_R4;
    float* expb  = ws + OFF_EXP;
    float* gatev = ws + OFF_GATE;
    float* part  = ws + OFF_PART;
    int*   eidx  = (int*)(ws + OFF_IDX);
    int*   slot  = (int*)(ws + OFF_SLOT);
    float* outp  = (float*)d_out;

    dim3 attn_grid(S_DIM / 64, NHEADS, B_DIM);

    // ===== Layer 0 (dense) =====
    launch_gemm(true, false, x, D_DIM, 0, l0_in_w, 1, D_DIM, 0, 1,
                l0_in_b, big, LD3, 0, NTOK, 3 * D_DIM, D_DIM, 1, stream);
    attention_kernel<<<attn_grid, 128, 0, stream>>>(big, r1);
    launch_gemm(true, false, r1, D_DIM, 0, l0_out_w, 1, D_DIM, 0, 1,
                l0_out_b, r2, D_DIM, 0, NTOK, D_DIM, D_DIM, 1, stream);
    add_ln_kernel<<<NTOK, 256, 0, stream>>>(r2, x, l0_n1_w, l0_n1_b, r3);
    launch_gemm(true, true, r3, D_DIM, 0, l0_l1_w, 1, D_DIM, 0, 1,
                l0_l1_b, big, H_DIM, 0, NTOK, H_DIM, D_DIM, 1, stream);
    launch_gemm(true, false, big, H_DIM, 0, l0_l2_w, 1, H_DIM, 0, 1,
                l0_l2_b, r2, D_DIM, 0, NTOK, D_DIM, H_DIM, 1, stream);
    add_ln_kernel<<<NTOK, 256, 0, stream>>>(r2, r3, l0_n2_w, l0_n2_b, r4);

    // ===== Layer 1 (switch MoE) =====
    launch_gemm(true, false, r4, D_DIM, 0, l1_in_w, 1, D_DIM, 0, 1,
                l1_in_b, big, LD3, 0, NTOK, 3 * D_DIM, D_DIM, 1, stream);
    attention_kernel<<<attn_grid, 128, 0, stream>>>(big, r1);
    launch_gemm(true, false, r1, D_DIM, 0, l1_out_w, 1, D_DIM, 0, 1,
                l1_out_b, r2, D_DIM, 0, NTOK, D_DIM, D_DIM, 1, stream);
    add_ln_kernel<<<NTOK, 256, 0, stream>>>(r2, r4, l1_n1_w, l1_n1_b, r3);

    // gating + capacity routing
    gate_kernel<<<NTOK, 256, 0, stream>>>(r3, l1_gw, gatev, eidx);
    fill_i32_kernel<<<32, 256, 0, stream>>>(slot, NTOK, -1);
    route_scan_kernel<<<1, 32, 0, stream>>>(eidx, slot);

    fill_f32_kernel<<<2048, 256, 0, stream>>>(
        expb, (long long)B_DIM * E_DIM * C_CAP * D_DIM, 0.0f);
    gather_kernel<<<NTOK, 256, 0, stream>>>(r3, eidx, slot, expb);

    // expert FFN (batched 32x: [512x1024] @ [1024x4096] -> relu -> @ [4096x1024])
    launch_gemm(false, true, expb, D_DIM, (long long)C_CAP * D_DIM,
                l1_w1, H_DIM, 1, (long long)D_DIM * H_DIM, E_DIM,
                nullptr, big, H_DIM, (long long)C_CAP * H_DIM,
                C_CAP, H_DIM, D_DIM, B_DIM * E_DIM, stream);
    launch_gemm(false, false, big, H_DIM, (long long)C_CAP * H_DIM,
                l1_w2, D_DIM, 1, (long long)H_DIM * D_DIM, E_DIM,
                nullptr, expb, D_DIM, (long long)C_CAP * D_DIM,
                C_CAP, D_DIM, H_DIM, B_DIM * E_DIM, stream);

    combine_kernel<<<NTOK, 256, 0, stream>>>(expb, gatev, eidx, slot, r2);
    add_ln_kernel<<<NTOK, 256, 0, stream>>>(r2, r3, l1_n2_w, l1_n2_b, r1);

    rowsum_kernel<<<NTOK, 256, 0, stream>>>(r1, part);
    final_reduce_kernel<<<1, 256, 0, stream>>>(part, outp);
}
